// MHAGCN_11845519802968
// MI455X (gfx1250) — matrix operand order repaired and checked by reading, not compiled, not run on hardware
//
#include <hip/hip_runtime.h>

// ---------------------------------------------------------------------------
// MHA-GCN for MI455X (gfx1250, wave32, WMMA).
// B=4096, C=6, L=1024, MID=512, HID=512, OUT=256, K=5 heads.
// Heavy GEMMs run as v_wmma_f32_16x16x32_bf16 with fp32 accumulation.
// 8 batches (48 rows = 3 WMMA row tiles) per workgroup; x block staged in LDS.
// ---------------------------------------------------------------------------

#define EPSV 1e-5f

typedef __attribute__((ext_vector_type(16))) __bf16 v16bf;
typedef __attribute__((ext_vector_type(8)))  float  v8f;

struct __attribute__((aligned(16))) U4 { unsigned x, y, z, w; };
struct __attribute__((aligned(8)))  U2 { unsigned x, y; };

union Frag { v16bf v; U4 q[2]; };

__device__ __forceinline__ unsigned short f2bf(float f) {
  unsigned u = __float_as_uint(f);
  u += 0x7FFFu + ((u >> 16) & 1u);           // round-to-nearest-even
  return (unsigned short)(u >> 16);
}
__device__ __forceinline__ float bf2f(unsigned short h) {
  return __uint_as_float(((unsigned)h) << 16);
}

__device__ __forceinline__ v8f wmma_bf16(v16bf a, v16bf b, v8f c) {
  return __builtin_amdgcn_wmma_f32_16x16x32_bf16(false, a, false, b,
                                                 (short)0, c, false, false);
}

// A fragment (16x32 bf16, rows x K) from row-major storage (LDS).
// lanes 0-15: row lr, K {0..7, 16..23}; lanes 16-31: row lr, K {8..15, 24..31}.
__device__ __forceinline__ v16bf frag_a(const unsigned short* base, int row0,
                                        int stride, int k0, int lr, int hb) {
  const unsigned short* p = base + (size_t)(row0 + lr) * stride + k0 + hb * 8;
  Frag u;
  u.q[0] = *(const U4*)p;
  u.q[1] = *(const U4*)(p + 16);
  return u.v;
}
// B fragment (32x16 bf16, K x cols) from N-major (transposed) weights:
// lanes 0-15: col lr, K {0..7, 16..23}; lanes 16-31: col lr, K {8..15, 24..31}.
__device__ __forceinline__ v16bf frag_b(const unsigned short* wt, int col,
                                        int stride, int k0, int lr, int hb) {
  const unsigned short* p = wt + (size_t)(col + lr) * stride + k0 + hb * 8;
  Frag u;
  u.q[0] = *(const U4*)p;
  u.q[1] = *(const U4*)(p + 16);
  return u.v;
}

// ---------------------------------------------------------------- converts
__global__ void k_cvt_x(const float4* __restrict__ x, U2* __restrict__ xb, int n4) {
  int i = blockIdx.x * blockDim.x + threadIdx.x;
  if (i >= n4) return;
  float4 v = x[i];
  U2 o;
  o.x = (unsigned)f2bf(v.x) | ((unsigned)f2bf(v.y) << 16);
  o.y = (unsigned)f2bf(v.z) | ((unsigned)f2bf(v.w) << 16);
  xb[i] = o;
}

// Wt  : 5632 cols x 1024 K  = [aW1 k0..k4 | aW2 k0..k4 | gW1], N-major bf16
// Wt2 : 256  cols x 512  K  = gW2^T, N-major bf16
__global__ void k_build_w(const float* __restrict__ aW1, const float* __restrict__ aW2,
                          const float* __restrict__ gW1, const float* __restrict__ gW2,
                          unsigned short* __restrict__ wt, unsigned short* __restrict__ wt2) {
  int i = blockIdx.x * blockDim.x + threadIdx.x;
  const int NW = 5632 * 1024;
  if (i < NW) {
    int col = i >> 10, kk = i & 1023;
    float v;
    if (col < 2560) {
      int k = col >> 9, m = col & 511;
      v = aW1[((size_t)k * 1024 + kk) * 512 + m];
    } else if (col < 5120) {
      int c2 = col - 2560, k = c2 >> 9, m = c2 & 511;
      v = aW2[((size_t)k * 1024 + kk) * 512 + m];
    } else {
      v = gW1[(size_t)kk * 512 + (col - 5120)];
    }
    wt[(size_t)col * 1024 + kk] = f2bf(v);
  } else {
    int i2 = i - NW;
    if (i2 < 256 * 512) {
      int col = i2 >> 9, kk = i2 & 511;
      wt2[(size_t)col * 512 + kk] = f2bf(gW2[(size_t)kk * 256 + col]);
    }
  }
}

__global__ void k_zero(float* __restrict__ p, int n) {
  int i = blockIdx.x * blockDim.x + threadIdx.x;
  if (i < n) p[i] = 0.f;
}

// ------------------------------------------------ big fused WMMA GEMM kernel
// Per WG: 8 batches (48 rows). q/kv projections + block-diagonal score
// reduction, and xg = x @ gW1 (bf16 out). 8 waves; wave owns (head, m-tile).
__global__ __launch_bounds__(256) void k_qkv_scores(
    const unsigned short* __restrict__ xb, const unsigned short* __restrict__ wt,
    const float* __restrict__ ab1, const float* __restrict__ ab2,
    float* __restrict__ scores, unsigned short* __restrict__ xg) {
  extern __shared__ char smem[];
  unsigned short* xs = (unsigned short*)smem;          // 48 x 1024 bf16 (96 KB)
  float* qbase = (float*)(smem + 48 * 1024 * 2);       // 8 waves x 48 x 16 f32
  float* kbase = qbase + 8 * 48 * 16;                  // 8 waves x 48 x 16 f32

  const int tid = threadIdx.x;
  const int wave = tid >> 5;
  const int lane = tid & 31;
  const int hb = lane >> 4;
  const int lr = lane & 15;
  const int batch0 = blockIdx.x * 8;

  {  // stage x rows into LDS (16B vector copies)
    const U4* src = (const U4*)(xb + (size_t)batch0 * 6 * 1024);
    U4* dst = (U4*)xs;
    for (int i = tid; i < 48 * 1024 / 8; i += 256) dst[i] = src[i];
  }
  __syncthreads();

  float* qw = qbase + wave * (48 * 16);
  float* kw = kbase + wave * (48 * 16);
  const v8f vzero = {0.f, 0.f, 0.f, 0.f, 0.f, 0.f, 0.f, 0.f};

  // ---- attention: q & kv tiles + incremental score accumulation ----
  for (int kh = 0; kh < 5; ++kh) {
    float sacc[9];
#pragma unroll
    for (int t = 0; t < 9; ++t) sacc[t] = 0.f;

    for (int mt = wave; mt < 32; mt += 8) {        // 4 m-tiles per wave
      const int colq = kh * 512 + mt * 16;
      const int colk = 2560 + kh * 512 + mt * 16;
      v8f qa[3], ka[3];
      qa[0] = qa[1] = qa[2] = vzero;
      ka[0] = ka[1] = ka[2] = vzero;
      __builtin_prefetch(wt + (size_t)(colq + lr) * 1024, 0, 1);
      __builtin_prefetch(wt + (size_t)(colk + lr) * 1024, 0, 1);
      for (int kt = 0; kt < 32; ++kt) {
        v16bf bq = frag_b(wt, colq, 1024, kt * 32, lr, hb);
        v16bf bk = frag_b(wt, colk, 1024, kt * 32, lr, hb);
#pragma unroll
        for (int rt = 0; rt < 3; ++rt) {
          v16bf a = frag_a(xs, rt * 16, 1024, kt * 32, lr, hb);
          qa[rt] = wmma_bf16(a, bq, qa[rt]);
          ka[rt] = wmma_bf16(a, bk, ka[rt]);
        }
      }
      // biases (per output column m), then spill 48x16 tiles to wave scratch
      const float b1v = ab1[kh * 512 + mt * 16 + lr];
      const float b2v = ab2[kh * 512 + mt * 16 + lr];
#pragma unroll
      for (int rt = 0; rt < 3; ++rt)
#pragma unroll
        for (int r = 0; r < 8; ++r) {
          int row = rt * 16 + r + hb * 8;   // C/D layout: M=r(+8), N=lane&15
          qw[row * 16 + lr] = qa[rt][r] + b1v;
          kw[row * 16 + lr] = ka[rt][r] + b2v;
        }
      asm volatile("s_wait_dscnt 0" ::: "memory");  // same-wave LDS RAW

      // block-diagonal 6x6 dot products: 288 pairs -> 9 per lane
#pragma unroll
      for (int t = 0; t < 9; ++t) {
        int idx = lane + 32 * t;
        int bl = idx / 36, rem = idx % 36;
        int ii = rem / 6, jj = rem % 6;
        const float4* qr = (const float4*)(qw + (bl * 6 + ii) * 16);
        const float4* kr = (const float4*)(kw + (bl * 6 + jj) * 16);
        float s = 0.f;
#pragma unroll
        for (int m = 0; m < 4; ++m) {
          float4 a = qr[m], c = kr[m];
          s += a.x * c.x + a.y * c.y + a.z * c.z + a.w * c.w;
        }
        sacc[t] += s;
      }
    }
    // flush partials (8 waves per WG target the same (b,kh,i,j) entries)
#pragma unroll
    for (int t = 0; t < 9; ++t) {
      int idx = lane + 32 * t;
      int bl = idx / 36, rem = idx % 36;
      int ii = rem / 6, jj = rem % 6;
      float* dstp = scores + ((size_t)(batch0 + bl) * 5 + kh) * 36 + ii * 6 + jj;
      __hip_atomic_fetch_add(dstp, sacc[t], __ATOMIC_RELAXED,
                             __HIP_MEMORY_SCOPE_AGENT);
    }
  }

  // ---- xg = x @ gW1 (bf16 out, 32 n-tiles over 8 waves) ----
  for (int nt = wave; nt < 32; nt += 8) {
    const int col = 5120 + nt * 16;
    v8f acc[3];
    acc[0] = acc[1] = acc[2] = vzero;
    __builtin_prefetch(wt + (size_t)(col + lr) * 1024, 0, 1);
    for (int kt = 0; kt < 32; ++kt) {
      v16bf b = frag_b(wt, col, 1024, kt * 32, lr, hb);
#pragma unroll
      for (int rt = 0; rt < 3; ++rt) {
        v16bf a = frag_a(xs, rt * 16, 1024, kt * 32, lr, hb);
        acc[rt] = wmma_bf16(a, b, acc[rt]);
      }
    }
#pragma unroll
    for (int rt = 0; rt < 3; ++rt)
#pragma unroll
      for (int r = 0; r < 8; ++r) {
        int row = rt * 16 + r + hb * 8;
        xg[((size_t)batch0 * 6 + row) * 512 + nt * 16 + lr] = f2bf(acc[rt][r]);
      }
  }
}

// -------------------------------------- per-row softmax / exchange / softmax
__global__ void k_attn(const float* __restrict__ scores, const float* __restrict__ linW,
                       const float* __restrict__ linb, float* __restrict__ A) {
  int t = blockIdx.x * blockDim.x + threadIdx.x;
  if (t >= 4096 * 6) return;
  int b = t / 6, i = t % 6;
  float cat[30];
#pragma unroll
  for (int kh = 0; kh < 5; ++kh) {
    const float* s = scores + ((size_t)b * 5 + kh) * 36 + i * 6;
    float mx = s[0];
#pragma unroll
    for (int j = 1; j < 6; ++j) mx = fmaxf(mx, s[j]);
    float e[6], sum = 0.f;
#pragma unroll
    for (int j = 0; j < 6; ++j) { e[j] = __expf(s[j] - mx); sum += e[j]; }
    float r = 1.f / sum;
#pragma unroll
    for (int j = 0; j < 6; ++j) cat[kh * 6 + j] = e[j] * r;
  }
  float lg[6];
#pragma unroll
  for (int j = 0; j < 6; ++j) lg[j] = linb[j];
  for (int p = 0; p < 30; ++p) {
    float c = cat[p];
#pragma unroll
    for (int j = 0; j < 6; ++j) lg[j] += c * linW[p * 6 + j];
  }
  float mx = lg[0];
#pragma unroll
  for (int j = 1; j < 6; ++j) mx = fmaxf(mx, lg[j]);
  float sum = 0.f;
#pragma unroll
  for (int j = 0; j < 6; ++j) { lg[j] = __expf(lg[j] - mx); sum += lg[j]; }
  float r = 1.f / sum;
#pragma unroll
  for (int j = 0; j < 6; ++j) A[(size_t)t * 6 + j] = lg[j] * r;
}

// ------------------------------------- h = A@xg + gb1, PReLU, BN1  -> bf16
__global__ void k_h(const float* __restrict__ A, const unsigned short* __restrict__ xg,
                    const float* __restrict__ gb1, const float* __restrict__ prelu_a,
                    const float* __restrict__ g1, const float* __restrict__ b1,
                    const float* __restrict__ m1, const float* __restrict__ v1,
                    unsigned short* __restrict__ h) {
  size_t idx = (size_t)blockIdx.x * blockDim.x + threadIdx.x;
  if (idx >= (size_t)24576 * 512) return;
  int m = (int)(idx & 511);
  size_t row = idx >> 9;
  size_t b = row / 6;
  int ci = (int)(row % 6);
  const float* Ar = A + (b * 6 + ci) * 6;
  const unsigned short* xr = xg + b * 6 * 512 + m;
  float s = 0.f;
#pragma unroll
  for (int d = 0; d < 6; ++d) s += Ar[d] * bf2f(xr[(size_t)d * 512]);
  s += gb1[m];
  float a = prelu_a[0];
  s = (s >= 0.f) ? s : a * s;
  float inv = g1[ci] * rsqrtf(v1[ci] + EPSV);
  s = (s - m1[ci]) * inv + b1[ci];
  h[idx] = f2bf(s);
}

// -------------------- hw = h@gW2 (WMMA) ; out = A@hw + gb2 ; BN2 -> d_out
__global__ __launch_bounds__(256) void k_out(
    const unsigned short* __restrict__ h, const unsigned short* __restrict__ wt2,
    const float* __restrict__ A, const float* __restrict__ gb2,
    const float* __restrict__ g2, const float* __restrict__ b2,
    const float* __restrict__ m2, const float* __restrict__ v2,
    float* __restrict__ out) {
  extern __shared__ char smem[];
  unsigned short* hs = (unsigned short*)smem;     // 48 x 512 bf16
  float* hw = (float*)(smem + 48 * 512 * 2);      // 48 x 256 f32

  const int tid = threadIdx.x, wave = tid >> 5, lane = tid & 31;
  const int hb = lane >> 4, lr = lane & 15;
  const int batch0 = blockIdx.x * 8;

  const U4* src = (const U4*)(h + (size_t)batch0 * 6 * 512);
  U4* dst = (U4*)hs;
  for (int i = tid; i < 48 * 512 / 8; i += 256) dst[i] = src[i];
  __syncthreads();

  const v8f vzero = {0.f, 0.f, 0.f, 0.f, 0.f, 0.f, 0.f, 0.f};
  for (int nt = wave; nt < 16; nt += 8) {
    v8f acc[3];
    acc[0] = acc[1] = acc[2] = vzero;
    for (int kt = 0; kt < 16; ++kt) {
      v16bf b = frag_b(wt2, nt * 16, 512, kt * 32, lr, hb);
#pragma unroll
      for (int rt = 0; rt < 3; ++rt) {
        v16bf a = frag_a(hs, rt * 16, 512, kt * 32, lr, hb);
        acc[rt] = wmma_bf16(a, b, acc[rt]);
      }
    }
#pragma unroll
    for (int rt = 0; rt < 3; ++rt)
#pragma unroll
      for (int r = 0; r < 8; ++r) {
        int row = rt * 16 + r + hb * 8;
        hw[row * 256 + nt * 16 + lr] = acc[rt][r];
      }
  }
  __syncthreads();

  const int j = tid;  // 256 output columns
  for (int g = 0; g < 48; ++g) {
    int b = g / 6, ci = g % 6;
    const float* Ar = A + ((size_t)(batch0 + b) * 6 + ci) * 6;
    float s = 0.f;
#pragma unroll
    for (int d = 0; d < 6; ++d) s += Ar[d] * hw[(b * 6 + d) * 256 + j];
    s += gb2[j];
    float inv = g2[ci] * rsqrtf(v2[ci] + EPSV);
    s = (s - m2[ci]) * inv + b2[ci];
    out[((size_t)(batch0 + b) * 6 + ci) * 256 + j] = s;
  }
}

// ---------------------------------------------------------------- launcher
static const size_t XB_OFF  = 0;                    // x bf16        50,331,648 B
static const size_t WT_OFF  = XB_OFF + 50331648;    // Wt bf16       11,534,336 B
static const size_t WT2_OFF = WT_OFF + 11534336;    // Wt2 bf16         262,144 B
static const size_t XG_OFF  = WT2_OFF + 262144;     // xg bf16       25,165,824 B
static const size_t SC_OFF  = XG_OFF + 25165824;    // scores f32     2,949,120 B
static const size_t A_OFF   = SC_OFF + 2949120;     // A f32            589,824 B
static const size_t H_OFF   = A_OFF + 589824;       // h bf16        25,165,824 B
                                                    // total ~116 MB of d_ws

extern "C" void kernel_launch(void* const* d_in, const int* in_sizes, int n_in,
                              void* d_out, int out_size, void* d_ws, size_t ws_size,
                              hipStream_t stream) {
  const float* x     = (const float*)d_in[0];
  const float* aW1   = (const float*)d_in[1];
  const float* ab1   = (const float*)d_in[2];
  const float* aW2   = (const float*)d_in[3];
  const float* ab2   = (const float*)d_in[4];
  const float* linW  = (const float*)d_in[5];
  const float* linb  = (const float*)d_in[6];
  const float* gW1   = (const float*)d_in[7];
  const float* gb1   = (const float*)d_in[8];
  const float* gW2   = (const float*)d_in[9];
  const float* gb2   = (const float*)d_in[10];
  const float* prelu = (const float*)d_in[11];
  const float* bn1g  = (const float*)d_in[12];
  const float* bn1b  = (const float*)d_in[13];
  const float* bn1m  = (const float*)d_in[14];
  const float* bn1v  = (const float*)d_in[15];
  const float* bn2g  = (const float*)d_in[16];
  const float* bn2b  = (const float*)d_in[17];
  const float* bn2m  = (const float*)d_in[18];
  const float* bn2v  = (const float*)d_in[19];
  float* out = (float*)d_out;

  char* ws = (char*)d_ws;
  unsigned short* xb   = (unsigned short*)(ws + XB_OFF);
  unsigned short* wt   = (unsigned short*)(ws + WT_OFF);
  unsigned short* wt2  = (unsigned short*)(ws + WT2_OFF);
  unsigned short* xg   = (unsigned short*)(ws + XG_OFF);
  float*          sc   = (float*)(ws + SC_OFF);
  float*          Amat = (float*)(ws + A_OFF);
  unsigned short* hbuf = (unsigned short*)(ws + H_OFF);

  // 1) x -> bf16 (vectorized x4)
  k_cvt_x<<<(6291456 + 255) / 256, 256, 0, stream>>>((const float4*)x, (U2*)xb, 6291456);
  // 2) transposed bf16 weight panels
  k_build_w<<<(5898240 + 255) / 256, 256, 0, stream>>>(aW1, aW2, gW1, gW2, wt, wt2);
  // 3) zero score accumulator (atomic target)
  k_zero<<<(737280 + 255) / 256, 256, 0, stream>>>(sc, 737280);
  // 4) fused q/kv/xg WMMA GEMM + score reduction (LDS: 96KB x + 48KB scratch)
  k_qkv_scores<<<512, 256, 147456, stream>>>(xb, wt, ab1, ab2, sc, xg);
  // 5) softmax -> exchange layer -> softmax = adjacency A
  k_attn<<<(24576 + 255) / 256, 256, 0, stream>>>(sc, linW, linb, Amat);
  // 6) h = A@xg + gb1, PReLU, BN1
  k_h<<<(12582912 + 255) / 256, 256, 0, stream>>>(Amat, xg, gb1, prelu,
                                                  bn1g, bn1b, bn1m, bn1v, hbuf);
  // 7) WMMA h@gW2 + fused A@hw + gb2 + BN2 epilogue
  k_out<<<512, 256, 98304, stream>>>(hbuf, wt2, Amat, gb2,
                                     bn2g, bn2b, bn2m, bn2v, out);
  (void)in_sizes; (void)n_in; (void)out_size; (void)ws_size;
}
